// LSTM_38757784879582
// MI455X (gfx1250) — matrix operand order repaired and checked by reading, not compiled, not run on hardware
//
#include <hip/hip_runtime.h>

#define T_STEPS 28
#define NIN     28
#define HID     64
#define NCLS    10
#define BT      16   // batch rows per workgroup
#define HP      72   // padded hmat row (halfs): 144B stride, 16B aligned, rotates banks
#define GP      24   // padded gate-LDS inner dim (floats): 96B stride, 16B aligned

typedef __attribute__((ext_vector_type(16))) _Float16 v16h;
typedef __attribute__((ext_vector_type(8)))  float    v8f;
typedef __attribute__((ext_vector_type(4))) unsigned  uint32x4;
typedef __attribute__((ext_vector_type(8)))  int      int32x8;
typedef __attribute__((ext_vector_type(4)))  int      int32x4;

__device__ __forceinline__ float fast_sigmoid(float x) {
    return __builtin_amdgcn_rcpf(1.0f + __expf(-x));
}
__device__ __forceinline__ float fast_tanh(float x) {
#if __has_builtin(__builtin_amdgcn_tanhf)
    return __builtin_amdgcn_tanhf(x);
#else
    return 1.0f - 2.0f * __builtin_amdgcn_rcpf(__expf(2.0f * x) + 1.0f);
#endif
}

// TDM: load a [16 rows x 28 f32] tile (row stride T*NIN f32) from global into LDS.
// D# group0: count=1, lds_addr, global_addr, type=2. group1: data_size=4B,
// tensor_dim0=28, tensor_dim1=16, tile_dim0=28, tile_dim1=16, dim0_stride=784.
__device__ __forceinline__ void tdm_load_xtile(const float* gsrc, unsigned lds_off) {
    unsigned long long ga = (unsigned long long)gsrc;
    uint32x4 g0;
    g0[0] = 1u;                                                  // count=1
    g0[1] = lds_off;                                             // lds_addr
    g0[2] = (unsigned)ga;                                        // global_addr[31:0]
    g0[3] = (unsigned)((ga >> 32) & 0x01ffffffull) | (2u << 30); // addr[56:32] | type=2
    int32x8 g1;
    g1[0] = 0x00020000;        // data_size=2 (4 bytes), no multicast/pad/iterate
    g1[1] = (NIN << 16);       // tensor_dim0[15:0] in bits[63:48]
    g1[2] = (BT  << 16);       // tensor_dim0 hi=0 | tensor_dim1[15:0]
    g1[3] = (NIN << 16);       // tensor_dim1 hi=0 | tile_dim0=28
    g1[4] = BT;                // tile_dim1=16, tile_dim2=0
    g1[5] = T_STEPS * NIN;     // tensor_dim0_stride low32 = 784
    g1[6] = 0;
    g1[7] = 0;
    int32x4 z4 = {0, 0, 0, 0};
    int32x8 z8 = {0, 0, 0, 0, 0, 0, 0, 0};
    __builtin_amdgcn_tensor_load_to_lds(g0, g1, z4, z4, z8, 0);
}

__global__ __launch_bounds__(128, 1)
void lstm_wmma_kernel(const float* __restrict__ x,
                      const float* __restrict__ W_ih,
                      const float* __restrict__ W_hh,
                      const float* __restrict__ b_ih,
                      const float* __restrict__ b_hh,
                      const float* __restrict__ W_out,
                      const float* __restrict__ b_out,
                      float* __restrict__ out)
{
    __shared__ float    xbuf[2][BT][NIN];   // double-buffered x_t tile (TDM dest)
    __shared__ float    gAll[4][HID][GP];   // [gate][hidden unit][batch row], col-major
    __shared__ _Float16 hmat[BT][HP];       // h_t as f16, row-major [m][u]

    const int tid = threadIdx.x;
    const int lid = tid & 31;
    const int wv  = tid >> 5;    // wave id = gate id (i,f,g,o)
    const int bn  = lid & 15;    // B-operand column / A-operand row within 16-tile
    const int bg  = lid >> 4;    // lane half-group

    const size_t xbase = (size_t)blockIdx.x * BT * T_STEPS * NIN;

    // h0 = 0
    for (int i = tid; i < BT * HP; i += 128)
        ((_Float16*)hmat)[i] = (_Float16)0.0f;

    // ---- Load weights into resident WMMA B-operand registers (f32 -> f16) ----
    // B-layout (16-bit, Kx16): lanes 0-15 hold K={0-7,16-23}, lanes 16-31 K={8-15,24-31};
    // half j of v16h = row K = (j&8)*2 + bg*8 + (j&7), column N = bn. K runs contiguously
    // along a W row (W[col][k]), so these are coalesced global loads.
    v16h  WihB[4];        // K=0..31 (28 real + pad)
    v16h  WhhB[4][2];     // K=0..63 in two K=32 chunks
    float bsum[4];
    #pragma unroll
    for (int n = 0; n < 4; ++n) {
        const int col = wv * HID + n * 16 + bn;       // gate-matrix row index
        const float* wr = W_ih + (size_t)col * NIN;
        #pragma unroll
        for (int j = 0; j < 16; ++j) {
            const int k = (j & 8) * 2 + bg * 8 + (j & 7);
            WihB[n][j] = (k < NIN) ? (_Float16)wr[k] : (_Float16)0.0f;
        }
        const float* wh = W_hh + (size_t)col * HID;
        #pragma unroll
        for (int c = 0; c < 2; ++c)
            #pragma unroll
            for (int j = 0; j < 16; ++j)
                WhhB[n][c][j] = (_Float16)wh[c * 32 + (j & 8) * 2 + bg * 8 + (j & 7)];
        bsum[n] = b_ih[col] + b_hh[col];
    }

    // Per-thread cell state: thread owns hidden unit eu, batch rows mb..mb+7
    const int eu = tid >> 1;
    const int mb = (tid & 1) * 8;
    float cst[8];
    #pragma unroll
    for (int j = 0; j < 8; ++j) cst[j] = 0.0f;

    // Prefetch x tile for t=0 via Tensor Data Mover (wave 0 only)
    if (wv == 0)
        tdm_load_xtile(x + xbase,
                       (unsigned)(unsigned long long)(uintptr_t)&xbuf[0][0][0]);

    for (int t = 0; t < T_STEPS; ++t) {
        // Pipeline: issue TDM for t+1, then wait for t's tile.
        if (wv == 0) {
            if (t + 1 < T_STEPS) {
                tdm_load_xtile(x + xbase + (size_t)(t + 1) * NIN,
                               (unsigned)(unsigned long long)(uintptr_t)&xbuf[(t + 1) & 1][0][0]);
                __builtin_amdgcn_s_wait_tensorcnt(1);
            } else {
                __builtin_amdgcn_s_wait_tensorcnt(0);
            }
        }
        __syncthreads();   // x_t ready; h_{t-1} in hmat; gAll free to overwrite

        // ---- Build A operands (16-bit A 16x32 layout):
        // lane group 0 (lanes 0-15): K in {0-7, 16-23}; group 1: K in {8-15, 24-31}
        v16h xA;
        {
            const float* xr = &xbuf[t & 1][bn][0];
            #pragma unroll
            for (int j = 0; j < 8; ++j)
                xA[j] = (_Float16)xr[bg * 8 + j];
            #pragma unroll
            for (int j = 0; j < 8; ++j) {
                const int k = 16 + bg * 8 + j;
                xA[8 + j] = (k < NIN) ? (_Float16)xr[k] : (_Float16)0.0f;
            }
        }
        v16h hA[2];
        #pragma unroll
        for (int c = 0; c < 2; ++c) {
            const int kb = c * 32;
            #pragma unroll
            for (int j = 0; j < 8; ++j) hA[c][j]     = hmat[bn][kb + bg * 8 + j];
            #pragma unroll
            for (int j = 0; j < 8; ++j) hA[c][8 + j] = hmat[bn][kb + 16 + bg * 8 + j];
        }

        // ---- gates = bias + x_t @ W_ih^T + h_{t-1} @ W_hh^T ----
        // Stage-major ordering: the 4 N-tile accumulators stay live across all
        // three K-stages, so every dependent WMMA->WMMA pair has 3 independent
        // WMMAs in between (covers the XDL RAW hazard window without stalls).
        v8f acc[4];
        #pragma unroll
        for (int n = 0; n < 4; ++n) {
            #pragma unroll
            for (int r = 0; r < 8; ++r) acc[n][r] = bsum[n];
        }
        #pragma unroll
        for (int n = 0; n < 4; ++n)
            acc[n] = __builtin_amdgcn_wmma_f32_16x16x32_f16(false, xA,    false, WihB[n],    (short)0, acc[n], false, false);
        #pragma unroll
        for (int n = 0; n < 4; ++n)
            acc[n] = __builtin_amdgcn_wmma_f32_16x16x32_f16(false, hA[0], false, WhhB[n][0], (short)0, acc[n], false, false);
        #pragma unroll
        for (int n = 0; n < 4; ++n)
            acc[n] = __builtin_amdgcn_wmma_f32_16x16x32_f16(false, hA[1], false, WhhB[n][1], (short)0, acc[n], false, false);

        // C/D layout: VGPR r, lane -> M = bg*8 + r, N = n*16 + bn.
        // Store column-major so each lane writes 8 contiguous f32 (2x b128).
        #pragma unroll
        for (int n = 0; n < 4; ++n) {
            float* p = &gAll[wv][n * 16 + bn][bg * 8];
            #pragma unroll
            for (int r = 0; r < 8; ++r) p[r] = acc[n][r];
        }
        __syncthreads();   // all 4 gate blocks visible

        // ---- elementwise LSTM cell; c stays in registers ----
        #pragma unroll
        for (int j = 0; j < 8; ++j) {
            const int m = mb + j;
            const float iv = fast_sigmoid(gAll[0][eu][m]);
            const float fv = fast_sigmoid(gAll[1][eu][m]);
            const float gv = fast_tanh   (gAll[2][eu][m]);
            const float ov = fast_sigmoid(gAll[3][eu][m]);
            cst[j] = fv * cst[j] + iv * gv;
            hmat[m][eu] = (_Float16)(ov * fast_tanh(cst[j]));
        }
        // loop-top barrier orders hmat writes vs next iteration's A-operand reads
    }
    __syncthreads();

    // ---- head: logits[16,10] = h_T @ W_out^T + b_out (tiny; VALU) ----
    for (int e = tid; e < BT * NCLS; e += 128) {
        const int m = e / NCLS, cls = e % NCLS;
        float s = b_out[cls];
        const float* wr = W_out + (size_t)cls * HID;
        #pragma unroll 8
        for (int u = 0; u < HID; ++u)
            s += (float)hmat[m][u] * wr[u];
        out[((size_t)blockIdx.x * BT + m) * NCLS + cls] = s;
    }
}

extern "C" void kernel_launch(void* const* d_in, const int* in_sizes, int n_in,
                              void* d_out, int out_size, void* d_ws, size_t ws_size,
                              hipStream_t stream) {
    const float* x     = (const float*)d_in[0];
    const float* W_ih  = (const float*)d_in[1];
    const float* W_hh  = (const float*)d_in[2];
    const float* b_ih  = (const float*)d_in[3];
    const float* b_hh  = (const float*)d_in[4];
    const float* W_out = (const float*)d_in[5];
    const float* b_out = (const float*)d_in[6];
    float* out = (float*)d_out;

    const int B = in_sizes[0] / (T_STEPS * NIN);   // 16384
    dim3 grid(B / BT), block(128);
    hipLaunchKernelGGL(lstm_wmma_kernel, grid, block, 0, stream,
                       x, W_ih, W_hh, b_ih, b_hh, W_out, b_out, out);
    (void)d_ws; (void)ws_size; (void)n_in; (void)out_size;
}
